// IlqlSampler_55181739819588
// MI455X (gfx1250) — matrix operand order repaired and checked by reading, not compiled, not run on hardware
//
#include <hip/hip_runtime.h>

// ---------------- problem constants ----------------
constexpr int  BB   = 128;
constexpr int  VV   = 128000;
constexpr int  DD   = 1024;
constexpr int  PLEN = 512;
constexpr int  OLEN = 128;

// GEMM tiling
constexpr int KC  = 32;     // K chunk per WMMA step (bf16 16x16x32)
constexpr int NT  = 128;    // vocab tile per workgroup (8 WMMA N-subtiles per wave)
constexpr int LDA = KC + 8; // LDS row stride (bank decorrelation)

// histogram config
constexpr int   NB  = 4096;
constexpr float WIN = 32.0f;
constexpr int   RT  = 512;  // threads in rowstats block

typedef __attribute__((ext_vector_type(16))) __bf16         v16bf;
typedef __attribute__((ext_vector_type(4)))  __bf16         v4bf;
typedef __attribute__((ext_vector_type(16))) unsigned short v16us;
typedef __attribute__((ext_vector_type(8)))  float          v8f;

struct RowRec { float m; float lnS4; int bstar; int pad; };

__device__ __forceinline__ void store_bf4(unsigned short* dst, float4 v) {
#if __has_builtin(__builtin_amdgcn_cvt_pk_bf16_f32)
    typedef __attribute__((ext_vector_type(2))) __bf16 v2bf;
    v2bf lo = __builtin_amdgcn_cvt_pk_bf16_f32(v.x, v.y);
    v2bf hi = __builtin_amdgcn_cvt_pk_bf16_f32(v.z, v.w);
    ((unsigned int*)dst)[0] = __builtin_bit_cast(unsigned int, lo);
    ((unsigned int*)dst)[1] = __builtin_bit_cast(unsigned int, hi);
#else
    // fptrunc f32->bf16 is RNE; native on gfx1250 (bf16 VALU present)
    v4bf b;
    b.x = (__bf16)v.x; b.y = (__bf16)v.y; b.z = (__bf16)v.z; b.w = (__bf16)v.w;
    *(v4bf*)dst = b;
#endif
}

// ---------------- kernel 0: zero the count matrix ----------------
__global__ void k_zero(unsigned int* __restrict__ counts, int n) {
    for (int i = blockIdx.x * blockDim.x + threadIdx.x; i < n;
         i += gridDim.x * blockDim.x)
        counts[i] = 0u;
}

// ---------------- kernel 1: scatter token occurrences ----------------
// word layout: bits[15:0] = output-token count, bit16 = prompt-seen flag
__global__ void k_scatter(const int* __restrict__ ptok, const int* __restrict__ otok,
                          unsigned int* __restrict__ counts) {
    int i   = blockIdx.x * blockDim.x + threadIdx.x;
    int np  = BB * PLEN;
    int tot = np + BB * OLEN;
    if (i >= tot) return;
    if (i < np) {
        int r = i / PLEN;
        int t = ptok[i];
        atomicOr(&counts[(size_t)r * VV + t], 0x10000u);
    } else {
        int j = i - np;
        int r = j / OLEN;
        int t = otok[j];
        atomicAdd(&counts[(size_t)r * VV + t], 1u);
    }
}

// ---------------- kernel 2: WMMA GEMM + bias + penalties + temperature ----------------
// logits[B,V] = hidden @ E^T + ebias ; then vLLM penalties ; /temps. Writes into out2.
// Double-buffered LDS (global loads overlap WMMA) + double-buffered B fragments
// (ds_load of fragment nt+1 overlaps WMMA of fragment nt).
__global__ __launch_bounds__(256)
void k_gemm(const float* __restrict__ hidden, const float* __restrict__ emb,
            const float* __restrict__ ebias,  const unsigned int* __restrict__ counts,
            const float* __restrict__ presp,  const float* __restrict__ freqp,
            const float* __restrict__ repp,   const float* __restrict__ temps,
            float* __restrict__ out2)
{
    __shared__ unsigned short As[2][BB][LDA];  // hidden tile, bf16 bits
    __shared__ unsigned short Bs[2][NT][LDA];  // embedding tile, bf16 bits

    const int tid   = threadIdx.x;
    const int lane  = tid & 31;
    const int wave  = tid >> 5;         // 8 waves: each owns 16 batch rows
    const int lr    = lane & 15;
    const int half  = lane >> 4;        // lane half select
    const int nBase = blockIdx.x * NT;

    v8f acc[8] = {};                    // 8 N-subtiles of 16x16 f32

    // per-thread cooperative-load coordinates (4 float4 each for A and B tiles)
    int lrow[4], lcol[4];
    #pragma unroll
    for (int j = 0; j < 4; ++j) {
        int idx = tid + j * 256;
        lrow[j] = idx >> 3;
        lcol[j] = (idx & 7) * 4;
    }

    float4 hreg[4], ereg[4];
    // prologue: stage K-chunk 0
    #pragma unroll
    for (int j = 0; j < 4; ++j) {
        hreg[j] = *(const float4*)(hidden + (size_t)lrow[j] * DD + lcol[j]);
        ereg[j] = *(const float4*)(emb + (size_t)(nBase + lrow[j]) * DD + lcol[j]);
    }
    #pragma unroll
    for (int j = 0; j < 4; ++j) {
        store_bf4(&As[0][lrow[j]][lcol[j]], hreg[j]);
        store_bf4(&Bs[0][lrow[j]][lcol[j]], ereg[j]);
    }
    __syncthreads();

    const int bkk = half * 16;          // B-fragment K offset within chunk

    for (int k0 = 0; k0 < DD; k0 += KC) {
        const int  cur  = (k0 / KC) & 1;
        const bool more = (k0 + KC) < DD;

        // issue next chunk's global loads first (latency hides under WMMAs)
        if (more) {
            #pragma unroll
            for (int j = 0; j < 4; ++j) {
                hreg[j] = *(const float4*)(hidden + (size_t)lrow[j] * DD + (k0 + KC) + lcol[j]);
                ereg[j] = *(const float4*)(emb + (size_t)(nBase + lrow[j]) * DD + (k0 + KC) + lcol[j]);
            }
        }

        // A fragment: row M = 16*wave + lr ; elems 0..15 = K kk..kk+15
        v16us au;
        {
            const int m  = wave * 16 + lr;
            const int kk = half * 16;
            #pragma unroll
            for (int j = 0; j < 8; ++j) au[j]     = As[cur][m][kk + j];
            #pragma unroll
            for (int j = 0; j < 8; ++j) au[8 + j] = As[cur][m][kk + 8 + j];
        }
        v16bf a = __builtin_bit_cast(v16bf, au);

        // software-pipelined B fragments: load nt+1 while WMMA nt runs
        v16us bu[2];
        #pragma unroll
        for (int j = 0; j < 16; ++j) bu[0][j] = Bs[cur][lr][bkk + j];

        #pragma unroll
        for (int nt = 0; nt < 8; ++nt) {
            if (nt < 7) {
                const int n = (nt + 1) * 16 + lr;
                #pragma unroll
                for (int j = 0; j < 16; ++j) bu[(nt + 1) & 1][j] = Bs[cur][n][bkk + j];
            }
            v16bf b = __builtin_bit_cast(v16bf, bu[nt & 1]);
            acc[nt] = __builtin_amdgcn_wmma_f32_16x16x32_bf16(
                false, a, false, b, (short)0, acc[nt], false, false);
        }

        // convert + store staged chunk into the other buffer
        if (more) {
            #pragma unroll
            for (int j = 0; j < 4; ++j) {
                store_bf4(&As[cur ^ 1][lrow[j]][lcol[j]], hreg[j]);
                store_bf4(&Bs[cur ^ 1][lrow[j]][lcol[j]], ereg[j]);
            }
        }
        __syncthreads();
    }

    // epilogue: C/D layout -> row = 16*wave + g + 8*half, col = nBase + 16*nt + lr
    #pragma unroll
    for (int nt = 0; nt < 8; ++nt) {
        const int   col = nBase + nt * 16 + lr;
        const float eb  = ebias[col];
        #pragma unroll
        for (int g = 0; g < 8; ++g) {
            const int row = wave * 16 + g + half * 8;
            float x = acc[nt][g] + eb;
            unsigned int cw = counts[(size_t)row * VV + col];
            float oc = (float)(cw & 0xFFFFu);
            float rp = (cw != 0u) ? repp[row] : 1.0f;
            x = (x > 0.0f) ? x / rp : x * rp;
            x -= freqp[row] * oc;
            x -= presp[row] * ((cw & 0xFFFFu) ? 1.0f : 0.0f);
            x /= temps[row];
            out2[(size_t)row * VV + col] = x;
        }
    }
}

// ---------------- kernel 3: per-row max, histogram, scan, cutoff, S4 ----------------
__global__ __launch_bounds__(RT)
void k_rowstats(const float* __restrict__ out2, const float* __restrict__ lbias,
                const float* __restrict__ top_ps, const int* __restrict__ top_ks,
                const float* __restrict__ min_ps, RowRec* __restrict__ recs)
{
    __shared__ float hc[NB];     // bin counts (exact in f32: < 2^24)
    __shared__ float he[NB];     // bin sums of exp(x - m)
    __shared__ float part[RT];
    __shared__ float part2[RT];
    __shared__ float sm, sS;
    __shared__ int   sb;

    const int r   = blockIdx.x;
    const int tid = threadIdx.x;
    const float* x  = out2  + (size_t)r * VV;
    const float* lb = lbias + (size_t)r * VV;

    // sweep 0: row max
    float lm = -3.0e38f;
    for (int i = tid; i < VV; i += RT) lm = fmaxf(lm, x[i]);
    part[tid] = lm;
    __syncthreads();
    for (int off = RT / 2; off > 0; off >>= 1) {
        if (tid < off) part[tid] = fmaxf(part[tid], part[tid + off]);
        __syncthreads();
    }
    if (tid == 0) sm = part[0];
    for (int i = tid; i < NB; i += RT) { hc[i] = 0.0f; he[i] = 0.0f; }
    __syncthreads();
    const float m     = sm;
    const float scale = (float)NB / WIN;

    // sweep 1: histogram via LDS float atomics
    for (int i = tid; i < VV; i += RT) {
        float v = x[i];
        int b = (int)((m - v) * scale);
        b = (b < 0) ? 0 : ((b > NB - 1) ? NB - 1 : b);
        atomicAdd(&hc[b], 1.0f);
        atomicAdd(&he[b], __expf(v - m));
    }
    __syncthreads();

    // scan to exclusive prefixes: thread owns bins [8t, 8t+8)
    float cloc[8], eloc[8];
    float rc = 0.0f, re = 0.0f;
    {
        const int base = tid * 8;
        #pragma unroll
        for (int j = 0; j < 8; ++j) {
            cloc[j] = rc; eloc[j] = re;
            rc += hc[base + j]; re += he[base + j];
        }
        part[tid] = rc; part2[tid] = re;
    }
    __syncthreads();
    for (int off = 1; off < RT; off <<= 1) {
        float vc = (tid >= off) ? part[tid - off]  : 0.0f;
        float ve = (tid >= off) ? part2[tid - off] : 0.0f;
        __syncthreads();
        part[tid] += vc; part2[tid] += ve;
        __syncthreads();
    }
    const float co = part[tid]  - rc;   // exclusive chunk offsets
    const float eo = part2[tid] - re;
    if (tid == RT - 1) sS = part2[RT - 1];   // total sum of exp(x - m)
    if (tid == 0) sb = 0;
    {
        const int base = tid * 8;
        #pragma unroll
        for (int j = 0; j < 8; ++j) {
            hc[base + j] = cloc[j] + co;   // exclusive count-above
            he[base + j] = eloc[j] + eo;   // exclusive exp-mass-above
        }
    }
    __syncthreads();

    // find last bin satisfying both (monotone) keep conditions
    {
        const float tk  = (float)top_ks[r];
        const float tpS = top_ps[r] * sS;
        int best = -1;
        const int base = tid * 8;
        #pragma unroll
        for (int j = 0; j < 8; ++j) {
            int b = base + j;
            if (hc[b] < tk && he[b] < tpS) best = b;
        }
        if (best >= 0) atomicMax(&sb, best);
    }
    __syncthreads();
    const int   bstar = sb;
    const float mp    = min_ps[r];
    const float lmp   = (mp > 0.0f) ? __logf(mp) : -3.0e38f;

    // sweep 2: S4 = sum over kept of exp(x - m + logit_bias)
    float s4 = 0.0f;
    for (int i = tid; i < VV; i += RT) {
        float v = x[i];
        int b = (int)((m - v) * scale);
        b = (b < 0) ? 0 : ((b > NB - 1) ? NB - 1 : b);
        if (b <= bstar && (v - m) >= lmp) s4 += __expf(v - m + lb[i]);
    }
    part[tid] = s4;
    __syncthreads();
    for (int off = RT / 2; off > 0; off >>= 1) {
        if (tid < off) part[tid] += part[tid + off];
        __syncthreads();
    }
    if (tid == 0) {
        RowRec rec; rec.m = m; rec.lnS4 = __logf(part[0]); rec.bstar = bstar; rec.pad = 0;
        recs[r] = rec;
    }
}

// ---------------- kernel 4: finalize probs / logprobs ----------------
__global__ __launch_bounds__(256)
void k_final(const float* __restrict__ lbias, const float* __restrict__ min_ps,
             const RowRec* __restrict__ recs, float* __restrict__ probs,
             float* __restrict__ out2)
{
    const int r = blockIdx.y;
    const int i = blockIdx.x * blockDim.x + threadIdx.x;
    if (i >= VV) return;
    const size_t idx = (size_t)r * VV + i;
    const RowRec rec = recs[r];

    float v = out2[idx];
    int b = (int)((rec.m - v) * ((float)NB / WIN));
    b = (b < 0) ? 0 : ((b > NB - 1) ? NB - 1 : b);
    const float mp  = min_ps[r];
    const float lmp = (mp > 0.0f) ? __logf(mp) : -3.0e38f;
    const bool keep = (b <= rec.bstar) && ((v - rec.m) >= lmp);

    float lp = keep ? (v - rec.m + lbias[idx] - rec.lnS4) : -__builtin_inff();
    probs[idx] = keep ? __expf(lp) : 0.0f;
    out2[idx]  = lp;
}

// ---------------- launcher ----------------
extern "C" void kernel_launch(void* const* d_in, const int* in_sizes, int n_in,
                              void* d_out, int out_size, void* d_ws, size_t ws_size,
                              hipStream_t stream) {
    (void)in_sizes; (void)n_in; (void)out_size; (void)ws_size;
    const float* hidden = (const float*)d_in[0];
    const float* emb    = (const float*)d_in[1];
    const float* ebias  = (const float*)d_in[2];
    const float* lbias  = (const float*)d_in[3];
    const int*   ptok   = (const int*)d_in[4];
    const int*   otok   = (const int*)d_in[5];
    const float* presp  = (const float*)d_in[6];
    const float* freqp  = (const float*)d_in[7];
    const float* repp   = (const float*)d_in[8];
    const float* temps  = (const float*)d_in[9];
    const float* topps  = (const float*)d_in[10];
    const int*   topks  = (const int*)d_in[11];
    const float* minps  = (const float*)d_in[12];

    float* probs = (float*)d_out;                 // output 0: probs [B,V]
    float* out2  = probs + (size_t)BB * VV;       // output 1: logprobs [B,V] (scratch for logits)

    unsigned int* counts = (unsigned int*)d_ws;                   // B*V u32 = 64 MB
    RowRec* recs = (RowRec*)((char*)d_ws + (size_t)BB * VV * 4);  // B records

    k_zero<<<4096, 256, 0, stream>>>(counts, BB * VV);
    k_scatter<<<(BB * (PLEN + OLEN) + 255) / 256, 256, 0, stream>>>(ptok, otok, counts);
    k_gemm<<<VV / NT, 256, 0, stream>>>(hidden, emb, ebias, counts,
                                        presp, freqp, repp, temps, out2);
    k_rowstats<<<BB, RT, 0, stream>>>(out2, lbias, topps, topks, minps, recs);
    k_final<<<dim3((VV + 255) / 256, BB), 256, 0, stream>>>(lbias, minps, recs, probs, out2);
}
